// JointNer_8229157339554
// MI455X (gfx1250) — matrix operand order repaired and checked by reading, not compiled, not run on hardware
//
#include <hip/hip_runtime.h>
#include <hip/hip_bf16.h>
#include <stdint.h>

// ---------------------------------------------------------------------------
// CDNA5 (gfx1250) WMMA types
// ---------------------------------------------------------------------------
typedef __attribute__((ext_vector_type(16))) __bf16 v16bf;
typedef __attribute__((ext_vector_type(8)))  float  v8f;

union FragU {
    v16bf v;
    uint4 q[2];
};

__device__ __forceinline__ __bf16 f2bf(float f) {
    uint32_t u = __builtin_bit_cast(uint32_t, f);
    uint32_t r = (u + 0x7FFFu + ((u >> 16) & 1u)) >> 16;   // round-to-nearest-even
    uint16_t s = (uint16_t)r;
    return __builtin_bit_cast(__bf16, s);
}

__device__ __forceinline__ uint32_t pack2bf(float a, float b) {
    uint16_t lo = __builtin_bit_cast(uint16_t, f2bf(a));
    uint16_t hi = __builtin_bit_cast(uint16_t, f2bf(b));
    return (uint32_t)lo | ((uint32_t)hi << 16);
}

__device__ __forceinline__ float sigmoidf_(float x) {
    return 1.0f / (1.0f + __expf(-x));
}

// ---------------------------------------------------------------------------
// Pack W [N,K] fp32 (row-major) into WMMA B-fragment layout for B = W^T [K,N],
// zero-padding columns N..Npad-1 so the GEMM inner loop needs no guards.
// Tile (nt, kt) of 32x16 (KxN); contiguous k order (A uses the same):
//   lane l: n = l%16, halves e=0..15 hold k = kt*32 + (l<16?0:16) + e.
// Per-lane fragment = 16 contiguous bf16 (32B) -> two global_load_b128 at use.
// Tile order: tile_id = nt*KT + kt (k-loop streams contiguously).
// ---------------------------------------------------------------------------
__global__ __launch_bounds__(256) void pack_b_kernel(
    const float* __restrict__ W, __bf16* __restrict__ dst,
    int N, int Npad, int K)
{
    int idx = blockIdx.x * blockDim.x + threadIdx.x;
    int total = Npad * K;
    if (idx >= total) return;
    int KT   = K >> 5;
    int tile = idx >> 9;         // 512 bf16 per tile
    int r    = idx & 511;
    int lane = r >> 4;
    int e    = r & 15;
    int nt   = tile / KT;
    int kt   = tile % KT;
    int n = nt * 16 + (lane & 15);
    int k = kt * 32 + ((lane < 16) ? 0 : 16) + e;
    float v = (n < N) ? W[(size_t)n * K + k] : 0.0f;
    dst[idx] = f2bf(v);
}

// fp32 -> bf16, 4 elements per thread (float4 load, 2x dword store)
__global__ __launch_bounds__(256) void f32_to_bf16_kernel(
    const float* __restrict__ src, __bf16* __restrict__ dst, int n4)
{
    int idx = blockIdx.x * blockDim.x + threadIdx.x;
    if (idx >= n4) return;
    float4 v = ((const float4*)src)[idx];
    uint2 o;
    o.x = pack2bf(v.x, v.y);
    o.y = pack2bf(v.z, v.w);
    ((uint2*)dst)[idx] = o;
}

// ---------------------------------------------------------------------------
// Tiled WMMA GEMM: C[M,N] (f32) = A_bf16[M,K] @ B(packed, Npad cols) + bias[n]
// Block = 512 threads = 16 waves. Block tile 64(M) x 256(N), K-step 64.
// Wave (wm = w&3, wn = w>>2): one 16-row M-tile x four 16-col N-tiles.
// A staged in LDS (64x64 bf16 per K-step), fragments via 2x ds_read_b128.
// B fragments read directly from packed global via 2x global_load_b128.
// Inner loop is branchless (B is zero-padded to the block tile).
// ---------------------------------------------------------------------------
__global__ __launch_bounds__(512) void gemm_bf16_kernel(
    const __bf16* __restrict__ A, const __bf16* __restrict__ Bp,
    const float* __restrict__ bias, float* __restrict__ C,
    int M, int N, int K)
{
    __shared__ __align__(16) __bf16 sA[64][64];   // 8 KB

    const int tid    = threadIdx.x;
    const int lane   = tid & 31;
    const int w      = tid >> 5;
    const int wm     = w & 3;
    const int wn     = w >> 2;
    const int m0     = blockIdx.y * 64;
    const int n0     = blockIdx.x * 256;
    const int KT     = K >> 5;                    // 32-wide k-tiles (B packing)
    const int KT64   = K >> 6;                    // 64-wide staging steps
    const int laneM  = lane & 15;
    const int hiHalf = lane >> 4;                 // 0 or 1

    v8f acc[4] = {};
    const int ntBase = (n0 >> 4) + wn * 4;

    for (int kt = 0; kt < KT64; ++kt) {
        // Cooperative A-tile load: 64 rows x 64 bf16 (16B per thread)
        {
            int row = tid >> 3;
            int seg = tid & 7;
            const uint4* src =
                (const uint4*)&A[(size_t)(m0 + row) * K + kt * 64 + seg * 8];
            *(uint4*)&sA[row][seg * 8] = *src;
        }
        __syncthreads();

#pragma unroll
        for (int sub = 0; sub < 2; ++sub) {
            const int ktg = kt * 2 + sub;         // 32-wide k-tile index
            // A fragment: halves 0-15 = k base..base+15 (base = hiHalf*16)
            FragU aF;
            {
                const __bf16* base = &sA[wm * 16 + laneM][sub * 32 + hiHalf * 16];
                aF.q[0] = *(const uint4*)(base);
                aF.q[1] = *(const uint4*)(base + 8);
            }
#pragma unroll
            for (int s = 0; s < 4; ++s) {
                FragU bF;
                const __bf16* bp =
                    Bp + ((size_t)(ntBase + s) * KT + ktg) * 512 + lane * 16;
                bF.q[0] = *(const uint4*)(bp);
                bF.q[1] = *(const uint4*)(bp + 8);
                acc[s] = __builtin_amdgcn_wmma_f32_16x16x32_bf16(
                    false, aF.v, false, bF.v, (short)0, acc[s], false, false);
            }
        }
        __syncthreads();
    }

    // Epilogue: C layout (lane 0-15: M=r, lane 16-31: M=r+8), add bias[n].
    // Store guard is wave-uniform (N is a multiple of 16).
#pragma unroll
    for (int s = 0; s < 4; ++s) {
        int n = (ntBase + s) * 16 + laneM;
        if (n < N) {
            float bv = bias ? bias[n] : 0.0f;
#pragma unroll
            for (int r = 0; r < 8; ++r) {
                int m = m0 + wm * 16 + (hiHalf ? (r + 8) : r);
                C[(size_t)m * N + n] = acc[s][r] + bv;
            }
        }
    }
}

// ---------------------------------------------------------------------------
// Persistent LSTM scan. Grid (4 batch-groups, 2 directions), 512 thr (16 waves).
// Per step: g[16,2048] = P[t] + h @ Whh^T via WMMA; h kept in LDS (16x512 bf16).
// Wave w owns hidden cols [32w, 32w+32) across ALL FOUR i/f/g/o chunks, so the
// pointwise LSTM update stays in-register (c state lives in VGPRs).
// ---------------------------------------------------------------------------
__global__ __launch_bounds__(512) void lstm_scan_kernel(
    const float* __restrict__ Pf, const float* __restrict__ Pb,
    const __bf16* __restrict__ WpF, const __bf16* __restrict__ WpB,
    float* __restrict__ Hout)
{
    const int T = 256;
    const int bg  = blockIdx.x;           // batch group: rows bg*16 .. bg*16+15
    const int dir = blockIdx.y;           // 0 = forward, 1 = backward
    const float*  Pd = dir ? Pb : Pf;     // [B*T, 2048] pre-activations (+bias)
    const __bf16* Wp = dir ? WpB : WpF;   // packed Whh^T fragments (K=512,N=2048)
    const int outOff = dir * 512;

    __shared__ __align__(16) __bf16 hbuf[16][512];   // 16 KB

    const int tid    = threadIdx.x;
    const int lane   = tid & 31;
    const int w      = tid >> 5;          // 0..15
    const int laneM  = lane & 15;
    const int hiHalf = lane >> 4;

    // h0 = 0
    {
        uint4 z = make_uint4(0u, 0u, 0u, 0u);
        ((uint4*)hbuf)[tid]       = z;
        ((uint4*)hbuf)[tid + 512] = z;
    }
    __syncthreads();

    float creg[2][8] = {};                // c0 = 0

    for (int step = 0; step < T; ++step) {
        const int t = dir ? (T - 1 - step) : step;

        v8f acc[4][2] = {};
#pragma unroll 1
        for (int kt = 0; kt < 16; ++kt) {           // K = 512
            FragU aF;
            const __bf16* abase = &hbuf[laneM][kt * 32 + hiHalf * 16];
            aF.q[0] = *(const uint4*)(abase);
            aF.q[1] = *(const uint4*)(abase + 8);
#pragma unroll
            for (int c = 0; c < 4; ++c) {           // i, f, g, o chunks
#pragma unroll
                for (int s = 0; s < 2; ++s) {
                    int ntile = c * 32 + 2 * w + s; // N-tile in [0,128)
                    FragU bF;
                    const __bf16* bp =
                        Wp + ((size_t)ntile * 16 + kt) * 512 + lane * 16;
                    bF.q[0] = *(const uint4*)(bp);
                    bF.q[1] = *(const uint4*)(bp + 8);
                    acc[c][s] = __builtin_amdgcn_wmma_f32_16x16x32_bf16(
                        false, aF.v, false, bF.v, (short)0, acc[c][s],
                        false, false);
                }
            }
        }

        // Pointwise LSTM update, fully in registers
        __bf16 hbf[2][8];
#pragma unroll
        for (int s = 0; s < 2; ++s) {
            int j = 32 * w + 16 * s + laneM;        // hidden col 0..511
#pragma unroll
            for (int r = 0; r < 8; ++r) {
                int mloc = hiHalf ? (r + 8) : r;
                int b    = bg * 16 + mloc;
                size_t prow = (size_t)(b * T + t) * 2048;
                float iv = acc[0][s][r] + Pd[prow + j];
                float fv = acc[1][s][r] + Pd[prow + 512 + j];
                float gv = acc[2][s][r] + Pd[prow + 1024 + j];
                float ov = acc[3][s][r] + Pd[prow + 1536 + j];
                float cc = sigmoidf_(fv) * creg[s][r]
                         + sigmoidf_(iv) * tanhf(gv);
                creg[s][r] = cc;
                float hh = sigmoidf_(ov) * tanhf(cc);
                hbf[s][r] = f2bf(hh);
                Hout[(size_t)(b * T + t) * 1024 + outOff + j] = hh;
            }
        }

        __syncthreads();   // everyone finished reading hbuf for this step
#pragma unroll
        for (int s = 0; s < 2; ++s) {
            int j = 32 * w + 16 * s + laneM;
#pragma unroll
            for (int r = 0; r < 8; ++r) {
                int mloc = hiHalf ? (r + 8) : r;
                hbuf[mloc][j] = hbf[s][r];
            }
        }
        __syncthreads();   // h(t) visible to all waves
    }
}

// out = lstm_out + sigmoid(gate) * highway (+ optional relu), emit bf16 input.
// 4 elements per thread: float4 loads, packed bf16x4 store.
__global__ __launch_bounds__(256) void combine_kernel(
    const float* __restrict__ Hout, const float* __restrict__ G,
    const float* __restrict__ Hw, __bf16* __restrict__ xout,
    int n4, int do_relu)
{
    int idx = blockIdx.x * blockDim.x + threadIdx.x;
    if (idx >= n4) return;
    float4 h = ((const float4*)Hout)[idx];
    float4 g = ((const float4*)G)[idx];
    float4 q = ((const float4*)Hw)[idx];
    float v0 = h.x + sigmoidf_(g.x) * q.x;
    float v1 = h.y + sigmoidf_(g.y) * q.y;
    float v2 = h.z + sigmoidf_(g.z) * q.z;
    float v3 = h.w + sigmoidf_(g.w) * q.w;
    if (do_relu) {
        v0 = fmaxf(v0, 0.0f); v1 = fmaxf(v1, 0.0f);
        v2 = fmaxf(v2, 0.0f); v3 = fmaxf(v3, 0.0f);
    }
    uint2 o;
    o.x = pack2bf(v0, v1);
    o.y = pack2bf(v2, v3);
    ((uint2*)xout)[idx] = o;
}

// crf[b,t,cur,prev] = E[b,t,cur] + trans[cur,prev]; 4 prev per thread (float4)
__global__ __launch_bounds__(256) void crf_kernel(
    const float* __restrict__ E, const float* __restrict__ trans,
    float* __restrict__ out, int total4)
{
    int idx = blockIdx.x * blockDim.x + threadIdx.x;
    if (idx >= total4) return;
    int base  = idx * 4;
    int prev0 = base & 31;
    int cur   = (base >> 5) & 31;
    int m     = base >> 10;
    float e = E[(size_t)m * 32 + cur];
    float4 tr = *(const float4*)&trans[cur * 32 + prev0];
    float4 o;
    o.x = e + tr.x; o.y = e + tr.y; o.z = e + tr.z; o.w = e + tr.w;
    ((float4*)out)[idx] = o;
}

// ---------------------------------------------------------------------------
// Host orchestration (graph-capture safe: launches only, all on `stream`)
// Input order: jax tree-flatten, dict keys sorted:
//   0 bert_out, 1 We, 2 be,
//   per layer l (base=3+10l): Wg, Wh, Whh_b, Whh_f, Wih_b, Wih_f, b_b, b_f, bg, bh
//   23 trans, 24 sent_lens (unused: full-length sequences)
// ---------------------------------------------------------------------------
extern "C" void kernel_launch(void* const* d_in, const int* in_sizes, int n_in,
                              void* d_out, int out_size, void* d_ws, size_t ws_size,
                              hipStream_t stream)
{
    (void)in_sizes; (void)ws_size; (void)out_size;
    if (n_in < 25) return;

    const int B = 64, T = 256, DIN = 768, H = 512;
    const int M = B * T;                 // 16384 tokens

    const float* bert = (const float*)d_in[0];
    const float* We   = (const float*)d_in[1];
    const float* be   = (const float*)d_in[2];
    struct Layer { const float *Wg,*Wh,*Whh_b,*Whh_f,*Wih_b,*Wih_f,*b_b,*b_f,*bg,*bh; };
    Layer L[2];
    for (int l = 0; l < 2; ++l) {
        const int base = 3 + l * 10;
        L[l].Wg    = (const float*)d_in[base + 0];
        L[l].Wh    = (const float*)d_in[base + 1];
        L[l].Whh_b = (const float*)d_in[base + 2];
        L[l].Whh_f = (const float*)d_in[base + 3];
        L[l].Wih_b = (const float*)d_in[base + 4];
        L[l].Wih_f = (const float*)d_in[base + 5];
        L[l].b_b   = (const float*)d_in[base + 6];
        L[l].b_f   = (const float*)d_in[base + 7];
        L[l].bg    = (const float*)d_in[base + 8];
        L[l].bh    = (const float*)d_in[base + 9];
    }
    const float* trans = (const float*)d_in[23];

    // Workspace carve-out
    char* p = (char*)d_ws;
    auto alloc = [&](size_t bytes) -> char* {
        char* r = p; p += (bytes + 255) & ~(size_t)255; return r;
    };
    __bf16* xA    = (__bf16*)alloc((size_t)M * 1024 * 2);
    __bf16* xB    = (__bf16*)alloc((size_t)M * 1024 * 2);
    __bf16* pWihF = (__bf16*)alloc((size_t)2048 * 1024 * 2);
    __bf16* pWihB = (__bf16*)alloc((size_t)2048 * 1024 * 2);
    __bf16* pWhhF = (__bf16*)alloc((size_t)2048 * 512 * 2);
    __bf16* pWhhB = (__bf16*)alloc((size_t)2048 * 512 * 2);
    __bf16* pWg   = (__bf16*)alloc((size_t)1024 * 1024 * 2);
    __bf16* pWh   = (__bf16*)alloc((size_t)1024 * 1024 * 2);
    __bf16* pWe   = (__bf16*)alloc((size_t)256 * 1024 * 2);   // N padded 32->256
    float*  Pf    = (float*)alloc((size_t)M * 2048 * 4);
    float*  Pb    = (float*)alloc((size_t)M * 2048 * 4);
    float*  Gg    = (float*)alloc((size_t)M * 1024 * 4);
    float*  Hw    = (float*)alloc((size_t)M * 1024 * 4);
    float*  Hout  = (float*)alloc((size_t)M * 1024 * 4);
    float*  E     = (float*)alloc((size_t)M * 32 * 4);

    // Layer-0 input -> bf16
    {
        int n4 = (M * DIN) / 4;
        f32_to_bf16_kernel<<<(n4 + 255) / 256, 256, 0, stream>>>(bert, xA, n4);
    }

    __bf16* xin = xA;
    __bf16* xnx = xB;
    for (int l = 0; l < 2; ++l) {
        const int K = (l == 0) ? DIN : 2 * H;   // 768 or 1024

        auto pack = [&](const float* Wsrc, __bf16* dst, int N_, int K_) {
            int n = N_ * K_;
            pack_b_kernel<<<(n + 255) / 256, 256, 0, stream>>>(Wsrc, dst, N_, N_, K_);
        };
        pack(L[l].Wih_f, pWihF, 4 * H, K);
        pack(L[l].Wih_b, pWihB, 4 * H, K);
        pack(L[l].Whh_f, pWhhF, 4 * H, H);
        pack(L[l].Whh_b, pWhhB, 4 * H, H);
        pack(L[l].Wg,    pWg,   2 * H, K);
        pack(L[l].Wh,    pWh,   2 * H, K);

        // Time-parallel projections (bias folded in)
        dim3 blk(512);
        dim3 g2048(2048 / 256, M / 64);
        gemm_bf16_kernel<<<g2048, blk, 0, stream>>>(xin, pWihF, L[l].b_f, Pf, M, 4 * H, K);
        gemm_bf16_kernel<<<g2048, blk, 0, stream>>>(xin, pWihB, L[l].b_b, Pb, M, 4 * H, K);
        dim3 g1024(1024 / 256, M / 64);
        gemm_bf16_kernel<<<g1024, blk, 0, stream>>>(xin, pWg, L[l].bg, Gg, M, 2 * H, K);
        gemm_bf16_kernel<<<g1024, blk, 0, stream>>>(xin, pWh, L[l].bh, Hw, M, 2 * H, K);

        // Sequential bidirectional recurrence (persistent, 8 workgroups)
        lstm_scan_kernel<<<dim3(4, 2), dim3(512), 0, stream>>>(Pf, Pb, pWhhF, pWhhB, Hout);

        // Highway combine (+relu after last layer -> feats), produce next bf16 x
        int n4 = (M * 1024) / 4;
        combine_kernel<<<(n4 + 255) / 256, 256, 0, stream>>>(
            Hout, Gg, Hw, xnx, n4, (l == 1) ? 1 : 0);

        __bf16* tmp = xin; xin = xnx; xnx = tmp;
    }

    // Emissions: E = feats @ We^T + be   (N=32, B zero-padded to 256 cols)
    {
        int n = 256 * 1024;
        pack_b_kernel<<<(n + 255) / 256, 256, 0, stream>>>(We, pWe, 32, 256, 1024);
        gemm_bf16_kernel<<<dim3(1, M / 64), dim3(512), 0, stream>>>(
            xin, pWe, be, E, M, 32, 1024);
    }

    // CRF score broadcast
    {
        int total4 = (M * 32 * 32) / 4;
        crf_kernel<<<(total4 + 255) / 256, 256, 0, stream>>>(
            E, trans, (float*)d_out, total4);
    }
}
